// TSModel_36524401885659
// MI455X (gfx1250) — matrix operand order repaired and checked by reading, not compiled, not run on hardware
//
#include <hip/hip_runtime.h>
#include <hip/hip_bf16.h>

typedef __attribute__((ext_vector_type(16))) _Float16 v16h;
typedef __attribute__((ext_vector_type(8)))  _Float16 v8h;
typedef __attribute__((ext_vector_type(8)))  float    v8f;

namespace {
constexpr int   T     = 2048;  // sequence length
constexpr int   H1    = 64;    // layer-1 hidden
constexpr int   NG    = 4 * H1;// 256 gate columns
constexpr int   HSTR  = 72;    // h1 LDS row stride in halves (row*36 mod 64 all distinct banks)
constexpr float LOG2E = 1.4426950408889634f;
}

__device__ __forceinline__ float fast_rcp(float x)  { return __builtin_amdgcn_rcpf(x); }
__device__ __forceinline__ float fast_exp2(float x) { return __builtin_amdgcn_exp2f(x); }
__device__ __forceinline__ float fast_sigmoid(float x) {
  return fast_rcp(1.0f + fast_exp2(-LOG2E * x));
}
__device__ __forceinline__ float fast_tanh(float x) {
  float xc = fminf(fmaxf(x, -30.0f), 30.0f);
  float e  = fast_exp2(-2.0f * LOG2E * xc);
  return (1.0f - e) * fast_rcp(1.0f + e);
}
// lane ^ 1 exchange: single ds_swizzle_b32 (SWAPX1: and=0x1f, or=0, xor=1)
__device__ __forceinline__ float lane_xor1(float v) {
  return __int_as_float(__builtin_amdgcn_ds_swizzle(__float_as_int(v), 0x041F));
}

// One workgroup = 16 batch rows, 8 wave32s, ONE barrier per timestep.
// Gate columns of W1 are permuted at load time so wave w owns all four gates
// (i,j,f,o) of hidden units 8w..8w+7:
//   tile nt=0, lane l15: column = 64*(l15&1) + 8w + (l15>>1)       (i | j)
//   tile nt=1, lane l15: column = 128 + 64*(l15&1) + 8w + (l15>>1) (f | o)
// After the 4 WMMAs + activations, one ds_swizzle (lane^1) gives each even
// lane (gi,gj,gf,go) for its unit -> c1/h1 update stays in-wave; h1 is
// ping-pong buffered in LDS (compile-time buffer index via 2x unroll) so one
// barrier per step orders everything. Layer 2 (transposed, 2 WMMAs on wave 0)
// is pipelined one step behind.
__global__ __launch_bounds__(256, 1)
void lstm2_wmma_kernel(const float* __restrict__ x,
                       const float* __restrict__ W1,
                       const float* __restrict__ b1,
                       const float* __restrict__ W2,
                       const float* __restrict__ b2,
                       float* __restrict__ out)
{
  __shared__ __align__(16) _Float16 h1s[2][16 * HSTR]; // ping-pong h1 state, f16
  __shared__ float xs[2][16];                          // ping-pong x_t

  const int tid  = threadIdx.x;
  const int lane = tid & 31;
  const int wv   = tid >> 5;    // wave 0..7
  const int lh   = lane >> 4;   // lane half (0/1)
  const int l15  = lane & 15;
  const int pr   = l15 & 1;     // parity: 0 -> {i,f}, 1 -> {j,o}
  const int un   = 8 * wv + (l15 >> 1);  // this lane-pair's hidden unit
  const int b0   = blockIdx.x * 16;

  // permuted gate columns for this lane's two tiles
  int gcol[2];
  gcol[0] = 64 * pr + un;          // i (even) | j (odd)
  gcol[1] = 128 + 64 * pr + un;    // f (even) | o (odd)

  // ---------- persistent layer-1 B-fragments (permuted W1 hidden rows, f16) ----
  v16h bf[2][2];
  float w0c[2], bc[2];
  #pragma unroll
  for (int nt = 0; nt < 2; ++nt) {
    w0c[nt] = W1[gcol[nt]];        // row 0: x weight (D == 1)
    bc[nt]  = b1[gcol[nt]];
    #pragma unroll
    for (int s = 0; s < 2; ++s) {
      v16h f;
      #pragma unroll
      for (int e = 0; e < 16; ++e) {
        const int k = 32 * s + ((e < 8) ? 0 : 16) + 8 * lh + (e & 7); // hidden row index 0..63
        f[e] = (_Float16)W1[(1 + k) * NG + gcol[nt]];
      }
      bf[nt][s] = f;
    }
  }

  // ---------- branchless per-lane activation constants (loop-invariant) -------
  // g = sc * rcp(1 + exp2(al2*z + be2)) + of   (log2e folded in)
  float al2[2], be2[2], scv[2], ofv[2];
  al2[0] = pr ? -2.0f * LOG2E : -LOG2E;  be2[0] = 0.0f;
  scv[0] = pr ? 2.0f : 1.0f;             ofv[0] = pr ? -1.0f : 0.0f;
  al2[1] = -LOG2E;                       be2[1] = pr ? 0.0f : -LOG2E; // forget bias 1.0
  scv[1] = 1.0f;                         ofv[1] = 0.0f;

  // ---------- persistent layer-2 A-fragment: W2^T (4x64, rows 4..15 zero) ------
  v16h a2[2];
  #pragma unroll
  for (int s = 0; s < 2; ++s) {
    v16h f;
    #pragma unroll
    for (int e = 0; e < 16; ++e) {
      const int v = e >> 1, pos = e & 1;
      const int k = 16 * lh + 2 * v + pos + 32 * s;
      f[e] = (l15 < 4) ? (_Float16)W2[k * 4 + l15] : (_Float16)0.0f;
    }
    a2[s] = f;
  }
  float b2v[4], w2hv[4];
  #pragma unroll
  for (int g = 0; g < 4; ++g) { b2v[g] = b2[g]; w2hv[g] = W2[H1 * 4 + g]; }

  // ---------- init state ----------
  float c1r[8];                 // c1 for (unit un, batches 8*lh .. 8*lh+7), even lanes
  #pragma unroll
  for (int v = 0; v < 8; ++v) c1r[v] = 0.f;
  for (int i = tid; i < 16 * HSTR; i += 256) h1s[0][i] = (_Float16)0.0f;
  if (tid < 16) xs[0][tid] = x[(size_t)(b0 + tid) * T];
  float c2r = 0.f, h2r = 0.f;   // layer-2 state, lane b of wave 0
  __syncthreads();

// One LSTM timestep; P is a compile-time literal (0/1) -> static LDS addressing.
#define LSTM_STEP(tv, P)                                                       \
  do {                                                                         \
    const int t_ = (tv);                                                       \
    v16h af[2];                                                                \
    _Pragma("unroll")                                                          \
    for (int s = 0; s < 2; ++s) {                                              \
      const _Float16* rp = &h1s[P][l15 * HSTR + 8 * lh + 32 * s];              \
      v8h lo = *(const v8h*)(rp);                                              \
      v8h hi = *(const v8h*)(rp + 16);                                         \
      v16h f;                                                                  \
      _Pragma("unroll")                                                        \
      for (int e = 0; e < 8; ++e) { f[e] = lo[e]; f[8 + e] = hi[e]; }          \
      af[s] = f;                                                               \
    }                                                                          \
    float xv[8];                                                               \
    _Pragma("unroll")                                                          \
    for (int v = 0; v < 8; ++v) xv[v] = xs[P][v + 8 * lh];                     \
    float ga[2][8];                                                            \
    _Pragma("unroll")                                                          \
    for (int nt = 0; nt < 2; ++nt) {                                           \
      v8f acc;                                                                 \
      _Pragma("unroll")                                                        \
      for (int v = 0; v < 8; ++v) acc[v] = xv[v] * w0c[nt] + bc[nt];           \
      acc = __builtin_amdgcn_wmma_f32_16x16x32_f16(false, af[0], false,        \
                bf[nt][0], (short)0, acc, false, false);                       \
      acc = __builtin_amdgcn_wmma_f32_16x16x32_f16(false, af[1], false,        \
                bf[nt][1], (short)0, acc, false, false);                       \
      _Pragma("unroll")                                                        \
      for (int v = 0; v < 8; ++v)                                              \
        ga[nt][v] = scv[nt] *                                                  \
            fast_rcp(1.0f + fast_exp2(acc[v] * al2[nt] + be2[nt])) + ofv[nt];  \
    }                                                                          \
    float gb0[8], gb1[8];                                                      \
    _Pragma("unroll")                                                          \
    for (int v = 0; v < 8; ++v) {                                              \
      gb0[v] = lane_xor1(ga[0][v]);                                            \
      gb1[v] = lane_xor1(ga[1][v]);                                            \
    }                                                                          \
    if (pr == 0) { /* even lanes: local = (i,f), remote = (j,o) */             \
      _Pragma("unroll")                                                        \
      for (int v = 0; v < 8; ++v) {                                            \
        const float c = c1r[v] * ga[1][v] + ga[0][v] * gb0[v];                 \
        c1r[v] = c;                                                            \
        const float h = fast_tanh(c) * gb1[v];                                 \
        h1s[1 - (P)][(v + 8 * lh) * HSTR + un] = (_Float16)h;                  \
      }                                                                        \
    }                                                                          \
    if (tid < 16 && t_ + 1 < T)                                                \
      xs[1 - (P)][tid] = x[(size_t)(b0 + tid) * T + (t_ + 1)];                 \
    if (wv == 0 && t_ > 0) { /* layer 2 for step t_-1 */                       \
      v8f acc2;                                                                \
      _Pragma("unroll")                                                        \
      for (int v = 0; v < 8; ++v)                                              \
        acc2[v] = (v < 4) ? (b2v[v] + h2r * w2hv[v]) : 0.0f;                   \
      _Pragma("unroll")                                                        \
      for (int s = 0; s < 2; ++s) {                                            \
        const _Float16* rp = &h1s[P][l15 * HSTR + 16 * lh + 32 * s];           \
        v8h lo = *(const v8h*)(rp);                                            \
        v8h hi = *(const v8h*)(rp + 8);                                        \
        v16h bfr;                                                              \
        _Pragma("unroll")                                                      \
        for (int e = 0; e < 8; ++e) { bfr[e] = lo[e]; bfr[8 + e] = hi[e]; }    \
        acc2 = __builtin_amdgcn_wmma_f32_16x16x32_f16(false, a2[s], false,     \
                   bfr, (short)0, acc2, false, false);                         \
      }                                                                        \
      if (lane < 16) {                                                         \
        const float i2 = fast_sigmoid(acc2[0]);                                \
        const float j2 = 2.0f * fast_sigmoid(2.0f * acc2[1]) - 1.0f;           \
        const float f2 = fast_sigmoid(acc2[2] + 1.0f);                         \
        const float o2 = fast_sigmoid(acc2[3]);                                \
        c2r = c2r * f2 + i2 * j2;                                              \
        h2r = fast_tanh(c2r) * o2;                                             \
        out[(size_t)(b0 + lane) * T + (t_ - 1)] = h2r;                         \
      }                                                                        \
    }                                                                          \
    __syncthreads();                                                           \
  } while (0)

  for (int t2 = 0; t2 < T; t2 += 2) {
    LSTM_STEP(t2, 0);
    LSTM_STEP(t2 + 1, 1);
  }
#undef LSTM_STEP

  // ============ epilogue: layer-2 for t = T-1 (h1(T-1) is in buffer T&1) ======
  if (wv == 0) {
    const int pe = T & 1;
    v8f acc2;
    #pragma unroll
    for (int v = 0; v < 8; ++v)
      acc2[v] = (v < 4) ? (b2v[v] + h2r * w2hv[v]) : 0.0f;
    #pragma unroll
    for (int s = 0; s < 2; ++s) {
      const _Float16* rp = &h1s[pe][l15 * HSTR + 16 * lh + 32 * s];
      v8h lo = *(const v8h*)(rp);
      v8h hi = *(const v8h*)(rp + 8);
      v16h bfr;
      #pragma unroll
      for (int e = 0; e < 8; ++e) { bfr[e] = lo[e]; bfr[8 + e] = hi[e]; }
      acc2 = __builtin_amdgcn_wmma_f32_16x16x32_f16(false, a2[s], false, bfr,
                                                    (short)0, acc2, false, false);
    }
    if (lane < 16) {
      const float i2 = fast_sigmoid(acc2[0]);
      const float j2 = 2.0f * fast_sigmoid(2.0f * acc2[1]) - 1.0f;
      const float f2 = fast_sigmoid(acc2[2] + 1.0f);
      const float o2 = fast_sigmoid(acc2[3]);
      c2r = c2r * f2 + i2 * j2;
      h2r = fast_tanh(c2r) * o2;
      out[(size_t)(b0 + lane) * T + (T - 1)] = h2r;
    }
  }
}

extern "C" void kernel_launch(void* const* d_in, const int* in_sizes, int n_in,
                              void* d_out, int out_size, void* d_ws, size_t ws_size,
                              hipStream_t stream) {
  const float* x  = (const float*)d_in[0];
  const float* W1 = (const float*)d_in[1];
  const float* b1 = (const float*)d_in[2];
  const float* W2 = (const float*)d_in[3];
  const float* b2 = (const float*)d_in[4];
  float* out = (float*)d_out;

  const int B       = in_sizes[0] / T;  // D == 1 -> 512
  const int nblocks = B / 16;           // 32 workgroups of 16 batch rows
  lstm2_wmma_kernel<<<dim3(nblocks), dim3(256), 0, stream>>>(x, W1, b1, W2, b2, out);

  (void)n_in; (void)out_size; (void)d_ws; (void)ws_size;
}
